// _ConvBlock4LessLayers_43018392436824
// MI455X (gfx1250) — matrix operand order repaired and checked by reading, not compiled, not run on hardware
//
#include <hip/hip_runtime.h>
#include <hip/hip_bf16.h>

// ---------------------------------------------------------------------------
// Types for CDNA5 WMMA (wave32)
// ---------------------------------------------------------------------------
typedef __bf16 bf16x8  __attribute__((ext_vector_type(8)));
typedef __bf16 v16bf   __attribute__((ext_vector_type(16)));
typedef float  v8f     __attribute__((ext_vector_type(8)));

static constexpr int CCH = 128;   // channels
static constexpr int LDA = 136;   // padded LDS row stride (elements); 272B rows: 16B aligned, bank-spread
static constexpr int MT  = 32;    // M rows per block (2 x 16-row WMMA subtiles per wave)
static constexpr int WMAT = CCH * LDA;          // elements per packed weight matrix (incl. pad)
static constexpr int WLAYER = 2 * WMAT;         // self + neigh per layer

// A fragment (16-bit A, 16x32): per-lane elems 0..7 = K = half*8+{0..7}, elems 8..15 = K = 16+half*8+{0..7}
__device__ inline v16bf ld_frag_a(const __bf16* p) {
    bf16x8 lo = *(const bf16x8*)(p);
    bf16x8 hi = *(const bf16x8*)(p + 16);
    return __builtin_shufflevector(lo, hi, 0,1,2,3,4,5,6,7,8,9,10,11,12,13,14,15);
}
// B fragment (16-bit B, 32x16): per-lane elems 0..7 = K = half*8+{0..7}, elems 8..15 = K = 16+half*8+{0..7}
__device__ inline v16bf ld_frag_b(const __bf16* p) {
    bf16x8 lo = *(const bf16x8*)(p);
    bf16x8 hi = *(const bf16x8*)(p + 16);
    return __builtin_shufflevector(lo, hi, 0,1,2,3,4,5,6,7,8,9,10,11,12,13,14,15);
}

// ---------------------------------------------------------------------------
// Zero fill
// ---------------------------------------------------------------------------
__global__ void zero_f32(float* __restrict__ p, int n) {
    int t = blockIdx.x * blockDim.x + threadIdx.x;
    if (t < n) p[t] = 0.0f;
}

// ---------------------------------------------------------------------------
// One-time weight repack: fp32 row-major W[k][n] -> bf16 transposed, padded
// LDS image WT[n*LDA + k] (pad region zeroed). 17408 elems per matrix.
// ---------------------------------------------------------------------------
__global__ void pack_weight(const float* __restrict__ w, __bf16* __restrict__ out) {
    int idx = blockIdx.x * blockDim.x + threadIdx.x;
    if (idx >= WMAT) return;
    int n = idx / LDA;
    int k = idx % LDA;
    out[idx] = (k < CCH) ? (__bf16)w[k * CCH + n] : (__bf16)0.0f;
}

// ---------------------------------------------------------------------------
// Edge-weighted scatter-add: out[dst[e], :] += attr[e] * feat[src[e], :]
// One wave (32 lanes) per edge; each lane owns 4 channels (float4 load +
// 4 f32 atomics). Edge index is wave-uniform -> scalarized index/attr loads.
// ---------------------------------------------------------------------------
__global__ void scatter_add(const float* __restrict__ feat,
                            const int*   __restrict__ src,
                            const int*   __restrict__ dst,
                            const float* __restrict__ attr,
                            float*       __restrict__ out,
                            int n_edges) {
    int t = blockIdx.x * blockDim.x + threadIdx.x;
    int e = t >> 5;                 // uniform across the wave32
    if (e >= n_edges) return;       // whole waves exit together
    e = __builtin_amdgcn_readfirstlane(e);
    const int lane = t & 31;
    const int s = src[e];
    const int d = dst[e];
    const float a = attr[e];
    const float4 v = *(const float4*)(feat + (size_t)s * CCH + lane * 4);
    float* o = out + (size_t)d * CCH + lane * 4;
    atomicAdd(o + 0, a * v.x);
    atomicAdd(o + 1, a * v.y);
    atomicAdd(o + 2, a * v.z);
    atomicAdd(o + 3, a * v.w);
}

// ---------------------------------------------------------------------------
// Fused dual-GEMM + bias + ReLU:
//   hout = relu(hin @ Wself + agg @ Wneigh + bias)
// Block = 256 threads = 8 waves; block computes a 32-row x 128-col tile.
// Wave w owns columns [16w, 16w+16) and two 16-row subtiles (16 WMMAs).
// Weights arrive pre-packed (bf16, transposed, padded) -> staged into LDS
// with plain b128 copies; activations converted fp32->bf16 on stage-in.
// ---------------------------------------------------------------------------
__global__ __launch_bounds__(256) void gemm_bias_relu(
        const float*  __restrict__ hin,
        const float*  __restrict__ agg,
        const __bf16* __restrict__ wpack,  // [2*WMAT]: Wself^T image, then Wneigh^T image
        const float*  __restrict__ bias,   // [128]
        float*        __restrict__ hout,
        int nrows) {
    __shared__ alignas(16) __bf16 sW[WLAYER];        // 69632 B
    __shared__ alignas(16) __bf16 sA[2 * MT * LDA];  // h tile then agg tile, 17408 B

    const int tid  = threadIdx.x;
    const int row0 = blockIdx.x * MT;

    // Stage packed weights: straight 16B copies (4352 uint4 chunks / 256 thr = 17 iters).
    {
        const uint4* wsrc = (const uint4*)wpack;
        uint4*       wdst = (uint4*)sW;
        for (int i = tid; i < (WLAYER * 2) / 16; i += 256) wdst[i] = wsrc[i];
    }
    // Stage the 32-row activation tiles (fp32 -> bf16, zero-padded past nrows).
    for (int idx = tid; idx < MT * CCH; idx += 256) {
        int m = idx >> 7;
        int c = idx & 127;
        int r = row0 + m;
        float vh = 0.0f, vg = 0.0f;
        if (r < nrows) {
            vh = hin[(size_t)r * CCH + c];
            vg = agg[(size_t)r * CCH + c];
        }
        sA[m * LDA + c]            = (__bf16)vh;
        sA[MT * LDA + m * LDA + c] = (__bf16)vg;
    }
    __syncthreads();

    const __bf16* sWs = sW;
    const __bf16* sWn = sW + WMAT;
    const __bf16* sAh = sA;
    const __bf16* sAg = sA + MT * LDA;

    const int wave = tid >> 5;
    const int lane = tid & 31;
    const int half = lane >> 4;     // 0: lanes 0-15, 1: lanes 16-31
    const int ml   = lane & 15;     // A row within subtile / B column within 16-col tile
    const int col  = wave * 16 + ml;

    v8f acc0 = {};
    v8f acc1 = {};
#pragma unroll
    for (int kb = 0; kb < CCH; kb += 32) {
        // B fragments shared by both M subtiles
        v16bf b_s = ld_frag_b(&sWs[col * LDA + kb + half * 8]);
        v16bf b_n = ld_frag_b(&sWn[col * LDA + kb + half * 8]);
        // subtile 0: rows [row0, row0+16)
        v16bf a_h0 = ld_frag_a(&sAh[ml * LDA + kb + half * 8]);
        v16bf a_g0 = ld_frag_a(&sAg[ml * LDA + kb + half * 8]);
        acc0 = __builtin_amdgcn_wmma_f32_16x16x32_bf16(false, a_h0, false, b_s,
                                                       (short)0, acc0, false, false);
        acc0 = __builtin_amdgcn_wmma_f32_16x16x32_bf16(false, a_g0, false, b_n,
                                                       (short)0, acc0, false, false);
        // subtile 1: rows [row0+16, row0+32)
        v16bf a_h1 = ld_frag_a(&sAh[(16 + ml) * LDA + kb + half * 8]);
        v16bf a_g1 = ld_frag_a(&sAg[(16 + ml) * LDA + kb + half * 8]);
        acc1 = __builtin_amdgcn_wmma_f32_16x16x32_bf16(false, a_h1, false, b_s,
                                                       (short)0, acc1, false, false);
        acc1 = __builtin_amdgcn_wmma_f32_16x16x32_bf16(false, a_g1, false, b_n,
                                                       (short)0, acc1, false, false);
    }

    // D layout: VGPR r holds M = r (lanes 0-15) and M = r+8 (lanes 16-31), N = lane&15.
    const float bv = bias[col];
#pragma unroll
    for (int r = 0; r < 8; ++r) {
        int m_out = r + half * 8;
        int row = row0 + m_out;
        if (row < nrows) {
            float v = acc0[r] + bv;
            hout[(size_t)row * CCH + col] = v > 0.0f ? v : 0.0f;
        }
        row += 16;
        if (row < nrows) {
            float v = acc1[r] + bv;
            hout[(size_t)row * CCH + col] = v > 0.0f ? v : 0.0f;
        }
    }
}

// ---------------------------------------------------------------------------
// Host-side orchestration (all on `stream`, graph-capture safe)
// ---------------------------------------------------------------------------
extern "C" void kernel_launch(void* const* d_in, const int* in_sizes, int n_in,
                              void* d_out, int out_size, void* d_ws, size_t ws_size,
                              hipStream_t stream) {
    const float* x          = (const float*)d_in[0];
    const int*   pool_src   = (const int*)  d_in[1];
    const int*   pool_dst   = (const int*)  d_in[2];
    const float* pool_attr  = (const float*)d_in[3];
    const int*   pp_src     = (const int*)  d_in[4];
    const int*   pp_dst     = (const int*)  d_in[5];
    const float* pp_attr    = (const float*)d_in[6];
    const int*   unpool_src = (const int*)  d_in[7];
    const int*   unpool_dst = (const int*)  d_in[8];
    const float* unpool_attr= (const float*)d_in[9];

    const float* Wself[4]  = { (const float*)d_in[10], (const float*)d_in[13],
                               (const float*)d_in[16], (const float*)d_in[19] };
    const float* Wneigh[4] = { (const float*)d_in[11], (const float*)d_in[14],
                               (const float*)d_in[17], (const float*)d_in[20] };
    const float* Bias[4]   = { (const float*)d_in[12], (const float*)d_in[15],
                               (const float*)d_in[18], (const float*)d_in[21] };

    const int E_POOL   = in_sizes[1];
    const int E_PP     = in_sizes[4];
    const int E_UNPOOL = in_sizes[7];
    const int NF       = in_sizes[0] / CCH;   // 100000
    const int NC       = 25000;               // module constant in the reference

    // Workspace layout: h | agg | h2 | packed bf16 weights (8 matrices)
    float* h   = (float*)d_ws;
    float* agg = h   + (size_t)NC * CCH;
    float* h2  = agg + (size_t)NC * CCH;
    __bf16* wpk = (__bf16*)(h2 + (size_t)NC * CCH);   // byte offset 38.4e6, 16B aligned
    float* out = (float*)d_out;

    const dim3 blk(256);
    auto nblk = [](long long n) { return dim3((unsigned)((n + 255) / 256)); };

    const long long nCoarseElems = (long long)NC * CCH;
    const long long nFineElems   = (long long)NF * CCH;

    // ---- One-time weight repack to bf16 transposed LDS image ----
    for (int l = 0; l < 4; ++l) {
        pack_weight<<<nblk(WMAT), blk, 0, stream>>>(Wself[l],  wpk + (size_t)l * WLAYER);
        pack_weight<<<nblk(WMAT), blk, 0, stream>>>(Wneigh[l], wpk + (size_t)l * WLAYER + WMAT);
    }

    // ---- Pooling: fine -> coarse ----
    zero_f32<<<nblk(nCoarseElems), blk, 0, stream>>>(h, (int)nCoarseElems);
    scatter_add<<<nblk((long long)E_POOL * 32), blk, 0, stream>>>(
        x, pool_src, pool_dst, pool_attr, h, E_POOL);

    // ---- 4 message-passing conv layers (ping-pong h <-> h2) ----
    float* bufs[2] = { h, h2 };
    const dim3 gemm_grid((NC + MT - 1) / MT);
    for (int l = 0; l < 4; ++l) {
        const float* hin  = bufs[l & 1];
        float*       hout = bufs[(l + 1) & 1];
        zero_f32<<<nblk(nCoarseElems), blk, 0, stream>>>(agg, (int)nCoarseElems);
        scatter_add<<<nblk((long long)E_PP * 32), blk, 0, stream>>>(
            hin, pp_src, pp_dst, pp_attr, agg, E_PP);
        gemm_bias_relu<<<gemm_grid, blk, 0, stream>>>(
            hin, agg, wpk + (size_t)l * WLAYER, Bias[l], hout, NC);
    }
    // after 4 layers the result is back in bufs[0] == h

    // ---- Unpooling: coarse -> fine ----
    zero_f32<<<nblk(nFineElems), blk, 0, stream>>>(out, (int)nFineElems);
    scatter_add<<<nblk((long long)E_UNPOOL * 32), blk, 0, stream>>>(
        h, unpool_src, unpool_dst, unpool_attr, out, E_UNPOOL);
}
